// NERClassifier_75101798138002
// MI455X (gfx1250) — matrix operand order repaired and checked by reading, not compiled, not run on hardware
//
#include <hip/hip_runtime.h>
#include <hip/hip_bf16.h>
#include <math.h>

typedef __attribute__((ext_vector_type(16))) _Float16 v16h;
typedef __attribute__((ext_vector_type(8)))  float    v8f;

// Problem constants
#define BB 64
#define SS 512
#define DD 768
#define HH 512
#define TT 37

#define KT1 (DD / 32)    // 24  K-tiles for GEMM1
#define NT1 (HH / 16)    // 32  N-tiles for GEMM1
#define KT2 (HH / 32)    // 16  K-tiles for GEMM2
#define NT2 3            // 48 cols (T=37 padded)
#define MTILES ((BB * SS) / 16)  // 2048
#define ACHUNKS (KT1 * 32)       // 768 32-byte A-fragment chunks per block

#if defined(__gfx1250__) && __has_builtin(__builtin_amdgcn_global_load_async_to_lds_b128)
#define USE_ASYNC_LDS 1
#else
#define USE_ASYNC_LDS 0
#endif

// Exact vector type the async builtin expects (per clang diagnostic):
// 'int __attribute__((__vector_size__(4 * sizeof(int))))'
typedef int v4i __attribute__((vector_size(16)));
typedef __attribute__((address_space(1))) v4i* gv4i_p;   // global b128 ptr
typedef __attribute__((address_space(3))) v4i* lv4i_p;   // LDS b128 ptr

// ---------------------------------------------------------------------------
// Packing kernels: fp32 operands -> fp16 WMMA fragments, each lane's 16 halfs
// contiguous (32 B) so GEMM loads are pure global_load_b128.
// ---------------------------------------------------------------------------
__global__ __launch_bounds__(256) void pack_x(
    const float* __restrict__ X, _Float16* __restrict__ Xp)
{
    const int gid  = blockIdx.x * 256 + threadIdx.x;
    const int frag = gid >> 5;                 // mTile*KT1 + kTile
    const int lane = gid & 31;
    if (frag >= MTILES * KT1) return;
    const int mTile = frag / KT1;
    const int kTile = frag % KT1;
    const int mlo = lane & 15, hi = lane >> 4;
    const float* row = X + (size_t)(mTile * 16 + mlo) * DD;
    const int ka = kTile * 32 + hi * 8;
    v16h a;
    #pragma unroll
    for (int j = 0; j < 8; ++j) {
        a[j]     = (_Float16)row[ka + j];
        a[j + 8] = (_Float16)row[ka + 16 + j];
    }
    *(v16h*)(Xp + ((size_t)frag * 32 + lane) * 16) = a;
}

__global__ __launch_bounds__(256) void pack_w1(
    const float* __restrict__ W1, _Float16* __restrict__ W1p)
{
    const int gid  = blockIdx.x * 256 + threadIdx.x;
    const int frag = gid >> 5;                 // kTile*NT1 + nTile
    const int lane = gid & 31;
    if (frag >= KT1 * NT1) return;
    const int kTile = frag / NT1;
    const int nTile = frag % NT1;
    const int mlo = lane & 15, hi = lane >> 4;
    const int kb = kTile * 32 + hi * 8;
    const int n  = nTile * 16 + mlo;
    v16h b;
    #pragma unroll
    for (int e = 0; e < 16; ++e)
        b[e] = (_Float16)W1[(size_t)(kb + (e & 7) + ((e >> 3) << 4)) * HH + n];
    *(v16h*)(W1p + ((size_t)frag * 32 + lane) * 16) = b;
}

__global__ __launch_bounds__(256) void pack_w2(
    const float* __restrict__ W2, _Float16* __restrict__ W2p)
{
    const int gid  = blockIdx.x * 256 + threadIdx.x;
    const int frag = gid >> 5;                 // kTile*NT2 + nTile
    const int lane = gid & 31;
    if (frag >= KT2 * NT2) return;
    const int kTile = frag / NT2;
    const int nTile = frag % NT2;
    const int mlo = lane & 15, hi = lane >> 4;
    const int kb = kTile * 32 + hi * 8;
    const int n  = nTile * 16 + mlo;
    v16h b;
    #pragma unroll
    for (int e = 0; e < 16; ++e)
        b[e] = (n < TT) ? (_Float16)W2[(size_t)(kb + (e & 7) + ((e >> 3) << 4)) * TT + n] : (_Float16)0.0f;
    *(v16h*)(W2p + ((size_t)frag * 32 + lane) * 16) = b;
}

// ---------------------------------------------------------------------------
// Fused MLP:  logits = relu(X @ W1 + b1) @ W2 + b2
// Block = 16 rows, 8 waves.  A fragments staged to LDS once via
// GLOBAL_LOAD_ASYNC_TO_LDS_B128 (ASYNCcnt).  B ping-pong double-buffered in
// registers (K-loop unrolled x2, no register copies) so the 8 B-loads of the
// next K-step are in flight while the 4 WMMAs of the current one execute.
// Phase-1 results scatter into LDS already in phase-2 A-fragment layout.
// ---------------------------------------------------------------------------
__global__ __launch_bounds__(256) void fused_mlp_kernel(
    const _Float16* __restrict__ Xp,    // packed A fragments
    const _Float16* __restrict__ W1p,   // packed B fragments (GEMM1)
    const float*    __restrict__ b1,
    const _Float16* __restrict__ W2p,   // packed B fragments (GEMM2)
    const float*    __restrict__ b2,
    float* __restrict__ logits)         // [B*S, T]
{
    __shared__ _Float16 aFrag[ACHUNKS * 16];    // 24 KB: block's A fragments
    __shared__ _Float16 hFrag[KT2 * 32 * 16];   // 16 KB: h in phase-2 A layout

    const int wave = threadIdx.x >> 5;
    const int lane = threadIdx.x & 31;
    const int mlo  = lane & 15;
    const int hi   = lane >> 4;
    const int rowBase = blockIdx.x * 16;

    // ---- Stage A fragments into LDS (once per block) ----
    {
        const v16h* Ag = (const v16h*)Xp + (size_t)blockIdx.x * ACHUNKS;
#if USE_ASYNC_LDS
        for (int i = threadIdx.x; i < ACHUNKS; i += 256) {
            const char* g = (const char*)(Ag + i);
            char*       l = (char*)&aFrag[(size_t)i * 16];
            __builtin_amdgcn_global_load_async_to_lds_b128(
                (gv4i_p)g, (lv4i_p)l, 0, 0);
            __builtin_amdgcn_global_load_async_to_lds_b128(
                (gv4i_p)(g + 16), (lv4i_p)(l + 16), 0, 0);
        }
#if __has_builtin(__builtin_amdgcn_s_wait_asynccnt)
        __builtin_amdgcn_s_wait_asynccnt(0);
#else
        asm volatile("s_wait_asynccnt 0" ::: "memory");
#endif
#else
        for (int i = threadIdx.x; i < ACHUNKS; i += 256)
            *(v16h*)&aFrag[(size_t)i * 16] = Ag[i];
#endif
        __syncthreads();
    }

    // ---------------- Phase 1 (ping-pong B buffers, unroll x2) -------------
    const v16h* Bbase = (const v16h*)W1p + lane;
    v8f acc[4] = {v8f{}, v8f{}, v8f{}, v8f{}};

    v16h bufA[4], bufB[4];
    #pragma unroll
    for (int t = 0; t < 4; ++t)
        bufA[t] = Bbase[(size_t)(wave * 4 + t) * 32];

    for (int kT = 0; kT < KT1; kT += 2) {
        // prefetch kT+1 into bufB  (kT+1 < KT1 always: KT1 even)
        #pragma unroll
        for (int t = 0; t < 4; ++t)
            bufB[t] = Bbase[(size_t)((kT + 1) * NT1 + wave * 4 + t) * 32];

        {   // compute kT with bufA
            const v16h a = *(const v16h*)&aFrag[((size_t)kT * 32 + lane) * 16];
            #pragma unroll
            for (int t = 0; t < 4; ++t)
                acc[t] = __builtin_amdgcn_wmma_f32_16x16x32_f16(
                    false, a, false, bufA[t], (short)0, acc[t], false, false);
        }

        // prefetch kT+2 into bufA
        if (kT + 2 < KT1) {
            #pragma unroll
            for (int t = 0; t < 4; ++t)
                bufA[t] = Bbase[(size_t)((kT + 2) * NT1 + wave * 4 + t) * 32];
        }

        {   // compute kT+1 with bufB
            const v16h a = *(const v16h*)&aFrag[((size_t)(kT + 1) * 32 + lane) * 16];
            #pragma unroll
            for (int t = 0; t < 4; ++t)
                acc[t] = __builtin_amdgcn_wmma_f32_16x16x32_f16(
                    false, a, false, bufB[t], (short)0, acc[t], false, false);
        }
    }

    // bias + ReLU -> scatter into phase-2 A-fragment layout in LDS
    #pragma unroll
    for (int t = 0; t < 4; ++t) {
        const int n = wave * 64 + t * 16 + mlo;     // h column == phase-2 K
        const float bias = b1[n];
        const int kT2  = n >> 5;
        const int sub  = n & 31;
        const int hi2  = (sub >> 3) & 1;
        const int e    = (sub & 7) + ((sub >> 4) << 3);
        #pragma unroll
        for (int r = 0; r < 8; ++r) {
            const int m = r + hi * 8;               // C layout: VGPR r -> row
            float v = acc[t][r] + bias;
            const int lane2 = m + hi2 * 16;
            hFrag[((kT2 * 32) + lane2) * 16 + e] =
                (_Float16)(v > 0.0f ? v : 0.0f);
        }
    }
    __syncthreads();

    // ---------------- Phase 2 (ping-pong, unroll x2) ----------------
    if (wave < NT2) {
        const int n = wave * 16 + mlo;              // output tag column
        v8f acc2 = v8f{};
        const v16h* B2 = (const v16h*)W2p + lane;

        v16h cA = B2[(size_t)wave * 32];
        v16h cB;
        for (int k2 = 0; k2 < KT2; k2 += 2) {
            cB = B2[(size_t)((k2 + 1) * NT2 + wave) * 32];   // KT2 even
            {
                const v16h a = *(const v16h*)&hFrag[(k2 * 32 + lane) * 16];
                acc2 = __builtin_amdgcn_wmma_f32_16x16x32_f16(
                    false, a, false, cA, (short)0, acc2, false, false);
            }
            if (k2 + 2 < KT2)
                cA = B2[(size_t)((k2 + 2) * NT2 + wave) * 32];
            {
                const v16h a = *(const v16h*)&hFrag[((k2 + 1) * 32 + lane) * 16];
                acc2 = __builtin_amdgcn_wmma_f32_16x16x32_f16(
                    false, a, false, cB, (short)0, acc2, false, false);
            }
        }
        if (n < TT) {
            const float bias = b2[n];
            #pragma unroll
            for (int r = 0; r < 8; ++r) {
                const int m = r + hi * 8;
                logits[(size_t)(rowBase + m) * TT + n] = acc2[r] + bias;
            }
        }
    }
}

// ---------------------------------------------------------------------------
// CRF negative log-likelihood per batch element.
// ---------------------------------------------------------------------------
__global__ __launch_bounds__(64) void crf_kernel(
    const float* __restrict__ logits,   // [B, S, T]
    const int*   __restrict__ mask,     // [B, S]
    const int*   __restrict__ labels,   // [B, S]
    const float* __restrict__ trans,    // [T, T]
    const float* __restrict__ start_t,  // [T]
    const float* __restrict__ end_t,    // [T]
    float* __restrict__ losses)         // [B]
{
    __shared__ float sTrans[TT * TT];
    __shared__ float sAlpha[TT];
    __shared__ float sRed[64];

    const int b   = blockIdx.x;
    const int tid = threadIdx.x;

    for (int i = tid; i < TT * TT; i += 64) sTrans[i] = trans[i];

    const float* lg = logits + (size_t)b * SS * TT;
    const int*   mk = mask   + (size_t)b * SS;
    const int*   lb = labels + (size_t)b * SS;

    float g = 0.0f;
    for (int s = tid; s < SS; s += 64) {
        const float mf = mk[s] ? 1.0f : 0.0f;
        g += mf * lg[s * TT + lb[s]];
        if (s >= 1) g += mf * trans[lb[s - 1] * TT + lb[s]];
    }
    sRed[tid] = g;
    __syncthreads();
    for (int off = 32; off > 0; off >>= 1) {
        if (tid < off) sRed[tid] += sRed[tid + off];
        __syncthreads();
    }

    if (tid < TT) sAlpha[tid] = start_t[tid] + lg[tid];
    __syncthreads();

    for (int s = 1; s < SS; ++s) {
        float aNew = 0.0f;
        const int m = mk[s];
        if (tid < TT) {
            float mx = -INFINITY;
            for (int i = 0; i < TT; ++i)
                mx = fmaxf(mx, sAlpha[i] + sTrans[i * TT + tid]);
            float sum = 0.0f;
            for (int i = 0; i < TT; ++i)
                sum += __expf(sAlpha[i] + sTrans[i * TT + tid] - mx);
            const float nv = mx + __logf(sum) + lg[s * TT + tid];
            aNew = m ? nv : sAlpha[tid];
        }
        __syncthreads();
        if (tid < TT) sAlpha[tid] = aNew;
        __syncthreads();
    }

    if (tid == 0) {
        float mx = -INFINITY;
        for (int t = 0; t < TT; ++t) mx = fmaxf(mx, sAlpha[t] + end_t[t]);
        float sum = 0.0f;
        for (int t = 0; t < TT; ++t) sum += __expf(sAlpha[t] + end_t[t] - mx);
        const float logZ = mx + __logf(sum);

        float gold = sRed[0] + start_t[lb[0]];
        int len = 0;
        for (int s = 0; s < SS; ++s) len += (mk[s] ? 1 : 0);
        gold += end_t[lb[len - 1]];

        losses[b] = logZ - gold;
    }
}

__global__ __launch_bounds__(64) void mean_kernel(
    const float* __restrict__ losses, float* __restrict__ out)
{
    __shared__ float s[64];
    s[threadIdx.x] = losses[threadIdx.x];
    __syncthreads();
    for (int off = 32; off > 0; off >>= 1) {
        if (threadIdx.x < off) s[threadIdx.x] += s[threadIdx.x + off];
        __syncthreads();
    }
    if (threadIdx.x == 0) out[0] = s[0] * (1.0f / (float)BB);
}

// ---------------------------------------------------------------------------
extern "C" void kernel_launch(void* const* d_in, const int* in_sizes, int n_in,
                              void* d_out, int out_size, void* d_ws, size_t ws_size,
                              hipStream_t stream) {
    (void)in_sizes; (void)n_in; (void)out_size; (void)ws_size;

    const float* word_reprs = (const float*)d_in[0];   // [B,S,D]
    const int*   word_mask  = (const int*)  d_in[1];   // [B,S]
    const int*   labels     = (const int*)  d_in[2];   // [B,S]
    const float* W1         = (const float*)d_in[3];   // [D,H]
    const float* b1         = (const float*)d_in[4];   // [H]
    const float* W2         = (const float*)d_in[5];   // [H,T]
    const float* b2         = (const float*)d_in[6];   // [T]
    const float* trans      = (const float*)d_in[7];   // [T,T]
    const float* start_t    = (const float*)d_in[8];   // [T]
    const float* end_t      = (const float*)d_in[9];   // [T]

    // Workspace layout (all 256-byte aligned)
    char* ws = (char*)d_ws;
    size_t off = 0;
    auto alloc = [&](size_t bytes) {
        char* p = ws + off;
        off = (off + bytes + 255) & ~(size_t)255;
        return p;
    };
    float*     logits = (float*)    alloc((size_t)BB * SS * TT * sizeof(float));
    float*     losses = (float*)    alloc((size_t)BB * sizeof(float));
    _Float16*  Xp     = (_Float16*) alloc((size_t)MTILES * KT1 * 512 * sizeof(_Float16));
    _Float16*  W1p    = (_Float16*) alloc((size_t)KT1 * NT1 * 512 * sizeof(_Float16));
    _Float16*  W2p    = (_Float16*) alloc((size_t)KT2 * NT2 * 512 * sizeof(_Float16));

    {   // packing
        const int tx  = MTILES * KT1 * 32;
        pack_x <<<(tx + 255) / 256, 256, 0, stream>>>(word_reprs, Xp);
        const int tw1 = KT1 * NT1 * 32;
        pack_w1<<<(tw1 + 255) / 256, 256, 0, stream>>>(W1, W1p);
        const int tw2 = KT2 * NT2 * 32;
        pack_w2<<<(tw2 + 255) / 256, 256, 0, stream>>>(W2, W2p);
    }

    fused_mlp_kernel<<<MTILES, 256, 0, stream>>>(Xp, W1p, b1, W2p, b2, logits);
    crf_kernel<<<BB, 64, 0, stream>>>(logits, word_mask, labels, trans, start_t,
                                      end_t, losses);
    mean_kernel<<<1, 64, 0, stream>>>(losses, (float*)d_out);
}
